// StreamingSVSModel_21234318311987
// MI455X (gfx1250) — matrix operand order repaired and checked by reading, not compiled, not run on hardware
//
#include <hip/hip_runtime.h>
#include <hip/hip_bf16.h>
#include <math.h>

// ---------------------------------------------------------------------------
// StreamingSVSModel forward for MI455X (gfx1250), wave32 + WMMA f16->f32.
// Audio decoder (~1.24 TFLOP of GEMMs) dominates: everything large runs
// through v_wmma_f32_16x16x32_f16 with double-buffered 128x128x32 LDS tiles.
// Attention K/V staging uses GLOBAL_LOAD_ASYNC_TO_LDS_B128 when available.
// ---------------------------------------------------------------------------

typedef __attribute__((ext_vector_type(16))) _Float16 v16h;
typedef __attribute__((ext_vector_type(8)))  float    v8f;

#define SVS_B   4
#define SVS_T   128
#define SVS_S   64
#define SVS_D   512
#define SVS_NH  8
#define SVS_DH  64
#define SVS_CS  1024

#if defined(__has_builtin)
#if __has_builtin(__builtin_amdgcn_global_load_async_to_lds_b128)
#define SVS_HAS_ASYNC_LDS 1
#endif
#if __has_builtin(__builtin_amdgcn_s_wait_asynccnt)
#define SVS_HAS_WAIT_ASYNC 1
#endif
#endif

#if defined(SVS_HAS_ASYNC_LDS)
// Builtin signature (from compiler diagnostic): first param is a pointer to
// int __vector(4). Build AS1/AS3-qualified int4-vector pointer types and cast
// through integers to avoid C++ address-space cast restrictions.
typedef int svs_v4i __attribute__((vector_size(16)));
typedef svs_v4i __attribute__((address_space(1)))* svs_gv4i;   // global
typedef svs_v4i __attribute__((address_space(3)))* svs_lv4i;   // LDS
#endif

// ===========================================================================
// WMMA GEMM:  C[m, n] = act( sum_k A[m,k]*W[n,k] + bias[n] ) + res[m,n]
//   A: [M,K] row-major f32 (activation), W: [N,K] row-major f32 (weights)
//   C written at C[m*ldc + n]  (ldc lets head GEMMs write into [..,K,CS] out)
//   act: 0=none, 1=silu, 2=gelu(exact)
//   TAIL=false requires K % 32 == 0 (all GEMMs except cond_w1 K=304):
//   no guards -> clean b128 global loads, no exec-mask gymnastics.
//   Double-buffered LDS: next tile's global loads issued before the WMMAs.
// ===========================================================================
template <bool TAIL>
__global__ __launch_bounds__(256) void k_gemm(
    const float* __restrict__ A, const float* __restrict__ W,
    const float* __restrict__ bias, const float* __restrict__ res, int ldres,
    float* __restrict__ C, int ldc, int M, int N, int K, int act)
{
  constexpr int BM = 128, BN = 128, BK = 32;
  __shared__ _Float16 As[2][BM * BK];   // [m][k] k-contiguous
  __shared__ _Float16 Bs[2][BN * BK];   // [n][k] k-contiguous
  const int tid  = threadIdx.x;
  const int lrow = tid >> 3;            // 0..31 staging row within 32-row pass
  const int kq   = (tid & 7) << 2;      // 0,4,..,28: 4 contiguous k per thread
  const int lane = tid & 31;
  const int wv   = tid >> 5;            // wave id 0..7
  const int wm   = wv & 3;              // 4 row groups of 32
  const int wn   = wv >> 2;             // 2 col groups of 64
  const int ll   = lane & 15;
  const int lh   = lane >> 4;

  v8f acc[2][4] = {};
  float ra[4][4], rb[4][4];             // staged registers for one tile

  auto load_tile = [&](int kt) {
    const int k0 = kt * BK;
    #pragma unroll
    for (int p = 0; p < 4; ++p) {
      const int m = p * 32 + lrow;
      const size_t ga = (size_t)(blockIdx.y * BM + m) * K + k0 + kq;
      const size_t gb = (size_t)(blockIdx.x * BN + m) * K + k0 + kq;
      if (!TAIL || (k0 + kq + 3) < K) {
        const float4 ta = *(const float4*)(A + ga);
        const float4 tb = *(const float4*)(W + gb);
        ra[p][0]=ta.x; ra[p][1]=ta.y; ra[p][2]=ta.z; ra[p][3]=ta.w;
        rb[p][0]=tb.x; rb[p][1]=tb.y; rb[p][2]=tb.z; rb[p][3]=tb.w;
      } else {
        #pragma unroll
        for (int j = 0; j < 4; ++j) {
          const bool ok = (k0 + kq + j) < K;
          ra[p][j] = ok ? A[ga + j] : 0.f;
          rb[p][j] = ok ? W[gb + j] : 0.f;
        }
      }
      if (k0 + BK < K) {                // -> global_prefetch_b8 (tile after next)
        __builtin_prefetch(A + ga + BK, 0, 3);
        __builtin_prefetch(W + gb + BK, 0, 3);
      }
    }
  };
  auto store_tile = [&](int buf) {
    #pragma unroll
    for (int p = 0; p < 4; ++p) {
      const int m = p * 32 + lrow;
      #pragma unroll
      for (int j = 0; j < 4; ++j) {
        As[buf][m * BK + kq + j] = (_Float16)ra[p][j];
        Bs[buf][m * BK + kq + j] = (_Float16)rb[p][j];
      }
    }
  };

  const int KT = TAIL ? (K + BK - 1) / BK : (K / BK);
  load_tile(0);
  store_tile(0);
  __syncthreads();

  for (int kt = 0; kt < KT; ++kt) {
    const int  buf  = kt & 1;
    const bool more = (kt + 1) < KT;
    if (more) load_tile(kt + 1);        // global loads in flight during WMMAs

    v16h af[2], bf[4];
    #pragma unroll
    for (int mt = 0; mt < 2; ++mt) {
      const int mrow = wm * 32 + mt * 16 + ll;
      #pragma unroll
      for (int v = 0; v < 8; ++v) {
        const int k = ((v < 4) ? (v * 2) : (16 + (v - 4) * 2)) + lh * 8;
        af[mt][2 * v]     = As[buf][mrow * BK + k];
        af[mt][2 * v + 1] = As[buf][mrow * BK + k + 1];
      }
    }
    #pragma unroll
    for (int nt = 0; nt < 4; ++nt) {
      const int nrow = wn * 64 + nt * 16 + ll;
      #pragma unroll
      for (int v = 0; v < 8; ++v) {
        const int k = ((v < 4) ? (v * 2) : (16 + (v - 4) * 2)) + lh * 8;
        bf[nt][2 * v]     = Bs[buf][nrow * BK + k];
        bf[nt][2 * v + 1] = Bs[buf][nrow * BK + k + 1];
      }
    }
    #pragma unroll
    for (int mt = 0; mt < 2; ++mt)
      #pragma unroll
      for (int nt = 0; nt < 4; ++nt)
        acc[mt][nt] = __builtin_amdgcn_wmma_f32_16x16x32_f16(
            false, af[mt], false, bf[nt], (short)0, acc[mt][nt], false, false);

    if (more) {
      __syncthreads();                  // everyone done reading buf^1 (prev iter)
      store_tile(buf ^ 1);
      __syncthreads();                  // stores visible before next compute
    }
  }

  // ---- epilogue: bias + activation + residual, strided store --------------
  #pragma unroll
  for (int mt = 0; mt < 2; ++mt) {
    #pragma unroll
    for (int nt = 0; nt < 4; ++nt) {
      const int n = blockIdx.x * BN + wn * 64 + nt * 16 + ll;
      #pragma unroll
      for (int v = 0; v < 8; ++v) {
        const int m = blockIdx.y * BM + wm * 32 + mt * 16 + lh * 8 + v;
        float val = acc[mt][nt][v];
        if (bias) val += bias[n];
        if (act == 1)      val = val / (1.f + __expf(-val));
        else if (act == 2) val = 0.5f * val * (1.f + erff(val * 0.70710678118654752f));
        if (res) val += res[(size_t)m * ldres + n];
        C[(size_t)m * ldc + n] = val;
      }
    }
  }
}

// ===========================================================================
// LayerNorm over rows of width D (block per row, 128 threads)
// ===========================================================================
__global__ __launch_bounds__(128) void k_layernorm(
    const float* __restrict__ x, float* __restrict__ y,
    const float* __restrict__ w, const float* __restrict__ b, int D)
{
  __shared__ float red[128];
  const size_t base = (size_t)blockIdx.x * D;
  float s = 0.f;
  for (int i = threadIdx.x; i < D; i += 128) s += x[base + i];
  red[threadIdx.x] = s; __syncthreads();
  for (int off = 64; off > 0; off >>= 1) {
    if (threadIdx.x < off) red[threadIdx.x] += red[threadIdx.x + off];
    __syncthreads();
  }
  const float mean = red[0] / D;
  __syncthreads();
  float v = 0.f;
  for (int i = threadIdx.x; i < D; i += 128) {
    const float d = x[base + i] - mean; v += d * d;
  }
  red[threadIdx.x] = v; __syncthreads();
  for (int off = 64; off > 0; off >>= 1) {
    if (threadIdx.x < off) red[threadIdx.x] += red[threadIdx.x + off];
    __syncthreads();
  }
  const float rstd = rsqrtf(red[0] / D + 1e-5f);
  __syncthreads();
  for (int i = threadIdx.x; i < D; i += 128)
    y[base + i] = (x[base + i] - mean) * rstd * w[i] + b[i];
}

// ===========================================================================
// Local-causal attention, thread-per-query online softmax.
// qkv: [R, S, 3*512] (q|k|v), out: [R, S, 512]. grid = R*8, block = S.
// K/V rows staged global->LDS via the CDNA5 async path (ASYNCcnt) when the
// toolchain exposes the builtin; otherwise plain copies.
// ===========================================================================
__global__ __launch_bounds__(128) void k_attn(
    const float* __restrict__ qkv, float* __restrict__ out,
    int S, int win, float scale)
{
  const int r = blockIdx.x >> 3;
  const int h = blockIdx.x & 7;
  extern __shared__ float sm[];
  float* Ks = sm;
  float* Vs = sm + (size_t)S * SVS_DH;
  const int s = threadIdx.x;
  const size_t rb = ((size_t)r * S + s) * (3 * SVS_D);

#if defined(SVS_HAS_ASYNC_LDS)
  {
    const float* gk = qkv + rb + SVS_D     + h * SVS_DH;
    const float* gv = qkv + rb + 2 * SVS_D + h * SVS_DH;
    float* lk = Ks + s * SVS_DH;
    float* lv = Vs + s * SVS_DH;
    #pragma unroll
    for (int j = 0; j < SVS_DH; j += 4) {
      __builtin_amdgcn_global_load_async_to_lds_b128(
          (svs_gv4i)(unsigned long long)(gk + j),
          (svs_lv4i)(unsigned int)(unsigned long long)(lk + j), 0, 0);
      __builtin_amdgcn_global_load_async_to_lds_b128(
          (svs_gv4i)(unsigned long long)(gv + j),
          (svs_lv4i)(unsigned int)(unsigned long long)(lv + j), 0, 0);
    }
#if defined(SVS_HAS_WAIT_ASYNC)
    __builtin_amdgcn_s_wait_asynccnt(0);
#else
    asm volatile("s_wait_asynccnt 0x0" ::: "memory");
#endif
  }
#else
  for (int j = 0; j < SVS_DH; ++j) {
    Ks[s * SVS_DH + j] = qkv[rb + SVS_D     + h * SVS_DH + j];
    Vs[s * SVS_DH + j] = qkv[rb + 2 * SVS_D + h * SVS_DH + j];
  }
#endif
  __syncthreads();

  float q[SVS_DH], o[SVS_DH];
  #pragma unroll
  for (int j = 0; j < SVS_DH; ++j) { q[j] = qkv[rb + h * SVS_DH + j]; o[j] = 0.f; }
  float m = -3.0e38f, l = 0.f;
  int k0 = s - win + 1; if (k0 < 0) k0 = 0;
  for (int k = k0; k <= s; ++k) {
    float d = 0.f;
    #pragma unroll
    for (int j = 0; j < SVS_DH; ++j) d += q[j] * Ks[k * SVS_DH + j];
    d *= scale;
    const float mn   = fmaxf(m, d);
    const float corr = __expf(m - mn);
    const float p    = __expf(d - mn);
    l = l * corr + p;
    #pragma unroll
    for (int j = 0; j < SVS_DH; ++j) o[j] = o[j] * corr + p * Vs[k * SVS_DH + j];
    m = mn;
  }
  const float inv = 1.f / l;
  const size_t ob = ((size_t)r * S + s) * SVS_D + h * SVS_DH;
  for (int j = 0; j < SVS_DH; ++j) out[ob + j] = o[j] * inv;
}

// ===========================================================================
// Elementwise / gather kernels
// ===========================================================================
__global__ void k_cond_embed(const int* __restrict__ note, const int* __restrict__ phon,
                             const int* __restrict__ slur, const int* __restrict__ pp,
                             const float* __restrict__ nw, const float* __restrict__ pw,
                             const float* __restrict__ sw, const float* __restrict__ ppw,
                             float* __restrict__ out, int total)
{
  const int i = blockIdx.x * blockDim.x + threadIdx.x;
  if (i >= total) return;
  const int row = i / 304, f = i % 304;
  float v;
  if (f < 128)       v = nw[note[row] * 128 + f];
  else if (f < 256)  v = pw[phon[row] * 128 + (f - 128)];
  else if (f < 272) { int sl = slur[row]; sl = sl < 0 ? 0 : (sl > 1 ? 1 : sl);
                      v = sw[sl * 16 + (f - 256)]; }
  else               v = ppw[pp[row] * 32 + (f - 272)];
  out[i] = v;
}

__global__ void k_gather_tok(const int* __restrict__ codes,
                             const float* __restrict__ emb,
                             float* __restrict__ out, int total)
{
  const int i = blockIdx.x * blockDim.x + threadIdx.x;
  if (i >= total) return;
  const int row = i >> 8, f = i & 255;           // width 256
  const int c = codes[(size_t)row * 2];          // codes[..,0]
  out[i] = emb[(size_t)c * 256 + f];
}

__global__ void k_pool(const float* __restrict__ tokemb, float* __restrict__ out)
{
  const int bt = blockIdx.x, f = threadIdx.x;    // block 256 = TOK width
  float s = 0.f;
  for (int j = 0; j < SVS_S; ++j) s += tokemb[((size_t)bt * SVS_S + j) * 256 + f];
  out[(size_t)bt * 256 + f] = s * (1.f / SVS_S);
}

__global__ void k_shift_prev(const float* __restrict__ prevu,
                             const float* __restrict__ bos,
                             float* __restrict__ out, int total)
{
  const int i = blockIdx.x * blockDim.x + threadIdx.x;
  if (i >= total) return;
  const int row = i >> 8, f = i & 255;
  const int t = row & (SVS_T - 1);
  out[i] = (t == 0) ? bos[f] : prevu[(size_t)(row - 1) * 256 + f];
}

__global__ void k_concat2(const float* __restrict__ A, int wa,
                          const float* __restrict__ Bp, int wb,
                          float* __restrict__ out, long total)
{
  const long i = (long)blockIdx.x * blockDim.x + threadIdx.x;
  if (i >= total) return;
  const int w = wa + wb;
  const long row = i / w;
  const int f = (int)(i - row * w);
  out[i] = (f < wa) ? A[row * wa + f] : Bp[row * wb + (f - wa)];
}

__global__ void k_a_init(const float* __restrict__ tok, const float* __restrict__ bos,
                         const float* __restrict__ fpos, const float* __restrict__ h,
                         float* __restrict__ a, long total)
{
  const long i = (long)blockIdx.x * blockDim.x + threadIdx.x;
  if (i >= total) return;
  const int  f   = (int)(i & (SVS_D - 1));
  const long row = i >> 9;                        // bt*S + s
  const int  s   = (int)(row & (SVS_S - 1));
  const long bt  = row >> 6;
  const float base = (s == 0) ? bos[f] : tok[(row - 1) * SVS_D + f];
  a[i] = base + fpos[s * SVS_D + f] + h[bt * SVS_D + f];
}

// ===========================================================================
// Host orchestration
// ===========================================================================
static inline void gemm(hipStream_t st, const float* A, const float* W,
                        const float* bias, const float* res, int ldres,
                        float* C, int ldc, int M, int N, int K, int act,
                        bool tail = false)
{
  dim3 g(N / 128, M / 128);
  if (tail)
    k_gemm<true><<<g, 256, 0, st>>>(A, W, bias, res, ldres, C, ldc, M, N, K, act);
  else
    k_gemm<false><<<g, 256, 0, st>>>(A, W, bias, res, ldres, C, ldc, M, N, K, act);
}

static void run_blocks(hipStream_t st, float* x, int rows, int Rseq, int Sseq, int win,
                       const float* const* P, int L,
                       float* lnbuf, float* qkvbuf, float* attnbuf, float* ffnbuf)
{
  for (int l = 0; l < L; ++l) {
    const float* ln1w = P[0]  + (size_t)l * 512;
    const float* ln1b = P[1]  + (size_t)l * 512;
    const float* inw  = P[2]  + (size_t)l * 1536 * 512;
    const float* inb  = P[3]  + (size_t)l * 1536;
    const float* outw = P[4]  + (size_t)l * 512 * 512;
    const float* outb = P[5]  + (size_t)l * 512;
    const float* ln2w = P[6]  + (size_t)l * 512;
    const float* ln2b = P[7]  + (size_t)l * 512;
    const float* f1w  = P[8]  + (size_t)l * 2048 * 512;
    const float* f1b  = P[9]  + (size_t)l * 2048;
    const float* f2w  = P[10] + (size_t)l * 512 * 2048;
    const float* f2b  = P[11] + (size_t)l * 512;

    k_layernorm<<<rows, 128, 0, st>>>(x, lnbuf, ln1w, ln1b, 512);
    gemm(st, lnbuf, inw, inb, nullptr, 0, qkvbuf, 1536, rows, 1536, 512, 0);
    k_attn<<<Rseq * SVS_NH, Sseq, (size_t)2 * Sseq * SVS_DH * sizeof(float), st>>>(
        qkvbuf, attnbuf, Sseq, win, 0.125f);
    gemm(st, attnbuf, outw, outb, x, 512, x, 512, rows, 512, 512, 0);
    k_layernorm<<<rows, 128, 0, st>>>(x, lnbuf, ln2w, ln2b, 512);
    gemm(st, lnbuf, f1w, f1b, nullptr, 0, ffnbuf, 2048, rows, 2048, 512, 2);
    gemm(st, ffnbuf, f2w, f2b, x, 512, x, 512, rows, 512, 2048, 0);
  }
}

extern "C" void kernel_launch(void* const* d_in, const int* in_sizes, int n_in,
                              void* d_out, int out_size, void* d_ws, size_t ws_size,
                              hipStream_t stream)
{
  (void)in_sizes; (void)n_in; (void)out_size; (void)ws_size;
  // ---- inputs (setup_inputs dict order) ----------------------------------
  const int*   codes      = (const int*)d_in[0];
  const int*   note_id    = (const int*)d_in[1];
  const int*   phoneme_id = (const int*)d_in[2];
  const int*   slur       = (const int*)d_in[3];
  const int*   pprog      = (const int*)d_in[4];
  const float* note_w     = (const float*)d_in[5];
  const float* ph_w       = (const float*)d_in[6];
  const float* slur_w     = (const float*)d_in[7];
  const float* ppw        = (const float*)d_in[8];
  const float* cond_w1    = (const float*)d_in[9];
  const float* cond_b1    = (const float*)d_in[10];
  const float* cond_w2    = (const float*)d_in[11];
  const float* cond_b2    = (const float*)d_in[12];
  const float* token_emb  = (const float*)d_in[13];
  const float* prev_w1    = (const float*)d_in[14];
  const float* prev_b1    = (const float*)d_in[15];
  const float* prev_w2    = (const float*)d_in[16];
  const float* prev_b2    = (const float*)d_in[17];
  const float* bos_step   = (const float*)d_in[18];
  const float* step_w1    = (const float*)d_in[19];
  const float* step_b1    = (const float*)d_in[20];
  const float* step_w2    = (const float*)d_in[21];
  const float* step_b2    = (const float*)d_in[22];
  const float* audio_in_w = (const float*)d_in[23];
  const float* audio_in_b = (const float*)d_in[24];
  const float* audio_bos  = (const float*)d_in[25];
  const float* frame_pos  = (const float*)d_in[26];
  const float* first_w    = (const float*)d_in[27];
  const float* first_b    = (const float*)d_in[28];
  const float* resc_w1    = (const float*)d_in[29];
  const float* resc_b1    = (const float*)d_in[30];
  const float* resc_w2    = (const float*)d_in[31];
  const float* resc_b2    = (const float*)d_in[32];
  const float* res_w      = (const float*)d_in[33];   // [1,1024,512]
  const float* res_b      = (const float*)d_in[34];   // [1,1024]
  const float* ctx_p[12];
  const float* aud_p[12];
  for (int j = 0; j < 12; ++j) ctx_p[j] = (const float*)d_in[35 + j];
  for (int j = 0; j < 12; ++j) aud_p[j] = (const float*)d_in[47 + j];

  // ---- workspace layout ---------------------------------------------------
  size_t off = 0;
  char* wsb = (char*)d_ws;
  auto F = [&](size_t nelem) {
    float* p = (float*)(wsb + off);
    off += ((nelem * sizeof(float) + 255) & ~(size_t)255);
    return p;
  };
  const long NBT  = SVS_B * SVS_T;            // 512
  const long NBTS = NBT * SVS_S;              // 32768
  float* cond_in  = F(NBT * 304);
  float* condh    = F(NBT * 256);
  float* cond     = F(NBT * 256);
  float* pooled   = F(NBT * 256);
  float* prevh    = F(NBT * 256);
  float* prevu    = F(NBT * 256);
  float* prevs    = F(NBT * 256);
  float* condprev = F(NBT * 512);
  float* hbuf1    = F(NBT * 512);
  float* h        = F(NBT * 512);
  float* ln_c     = F(NBT * 512);
  float* qkv_c    = F(NBT * 1536);
  float* attn_c   = F(NBT * 512);
  float* ffn_c    = F(NBT * 2048);
  float* tokemb   = F((size_t)NBTS * 256);
  float* tok      = F((size_t)NBTS * 512);
  float* a        = F((size_t)NBTS * 512);
  float* ln_a     = F((size_t)NBTS * 512);   // also reused as rh hidden
  float* attn_a   = F((size_t)NBTS * 512);   // also reused as rh
  float* big      = F((size_t)NBTS * 2048);  // qkv / ffn / concat (reused)
  float* out      = (float*)d_out;

  // ---- condition encoder --------------------------------------------------
  {
    const int total = (int)(NBT * 304);
    k_cond_embed<<<(total + 255) / 256, 256, 0, stream>>>(
        note_id, phoneme_id, slur, pprog, note_w, ph_w, slur_w, ppw, cond_in, total);
  }
  gemm(stream, cond_in, cond_w1, cond_b1, nullptr, 0, condh, 256, (int)NBT, 256, 304, 1,
       /*tail=*/true);
  gemm(stream, condh,   cond_w2, cond_b2, nullptr, 0, cond,  256, (int)NBT, 256, 256, 0);

  // ---- token embedding gather (shared by pooled + tok) --------------------
  {
    const int total = (int)(NBTS * 256);
    k_gather_tok<<<(total + 255) / 256, 256, 0, stream>>>(codes, token_emb, tokemb, total);
  }
  k_pool<<<(int)NBT, 256, 0, stream>>>(tokemb, pooled);
  gemm(stream, pooled, prev_w1, prev_b1, nullptr, 0, prevh, 256, (int)NBT, 256, 256, 1);
  gemm(stream, prevh,  prev_w2, prev_b2, nullptr, 0, prevu, 256, (int)NBT, 256, 256, 0);
  {
    const int total = (int)(NBT * 256);
    k_shift_prev<<<(total + 255) / 256, 256, 0, stream>>>(prevu, bos_step, prevs, total);
  }

  // ---- step MLP + context transformer (T=128, window 32) ------------------
  {
    const long total = NBT * 512;
    k_concat2<<<(int)((total + 255) / 256), 256, 0, stream>>>(cond, 256, prevs, 256, condprev, total);
  }
  gemm(stream, condprev, step_w1, step_b1, nullptr, 0, hbuf1, 512, (int)NBT, 512, 512, 1);
  gemm(stream, hbuf1,    step_w2, step_b2, nullptr, 0, h,     512, (int)NBT, 512, 512, 0);
  run_blocks(stream, h, (int)NBT, SVS_B, SVS_T, 32, ctx_p, 4, ln_c, qkv_c, attn_c, ffn_c);

  // ---- audio decoder input ------------------------------------------------
  gemm(stream, tokemb, audio_in_w, audio_in_b, nullptr, 0, tok, 512, (int)NBTS, 512, 256, 0);
  {
    const long total = NBTS * 512;
    k_a_init<<<(int)((total + 255) / 256), 256, 0, stream>>>(tok, audio_bos, frame_pos, h, a, total);
  }
  run_blocks(stream, a, (int)NBTS, (int)NBT, SVS_S, SVS_S, aud_p, 6, ln_a, big, attn_a, big);

  // ---- heads: out layout [B,T,S,K=2,CS=1024] -> ldc = 2048 ----------------
  gemm(stream, a, first_w, first_b, nullptr, 0, out, 2048, (int)NBTS, SVS_CS, 512, 0);
  {
    const long total = NBTS * 1024;
    k_concat2<<<(int)((total + 255) / 256), 256, 0, stream>>>(a, 512, tok, 512, big, total);
  }
  gemm(stream, big,    resc_w1, resc_b1, nullptr, 0, ln_a,   512, (int)NBTS, 512, 1024, 1);
  gemm(stream, ln_a,   resc_w2, resc_b2, nullptr, 0, attn_a, 512, (int)NBTS, 512, 512, 0);
  gemm(stream, attn_a, res_w,   res_b,   nullptr, 0, out + SVS_CS, 2048, (int)NBTS, SVS_CS, 512, 0);
}
